// SSMModel_36541581754488
// MI455X (gfx1250) — matrix operand order repaired and checked by reading, not compiled, not run on hardware
//
#include <hip/hip_runtime.h>

// ---------------------------------------------------------------------------
// SSM scan model for MI455X (gfx1250, wave32, WMMA + TDM).
//   bx = x @ B_w^T + B_b                (memory-bound GEMM, bf16 WMMA)
//   state_{t+1} = relu(state_t @ A_w^T + A_b + bx_t)   (latency-bound scan:
//       one workgroup / 16 waves; state double-buffered in LDS; A_w bf16
//       held *in VGPRs* across all 4096 steps; bx_t slices streamed ahead
//       into LDS by the Tensor Data Mover under TENSORcnt)
//   out = state_T @ C_w^T + C_b         (tiny GEMM)
// ---------------------------------------------------------------------------

typedef __attribute__((ext_vector_type(16))) __bf16 v16bf;
typedef __attribute__((ext_vector_type(8)))  __bf16 bf16x8;
typedef __attribute__((ext_vector_type(8)))  float  v8f;
typedef __attribute__((ext_vector_type(4)))  unsigned int uint32x4;
typedef __attribute__((ext_vector_type(8)))  int int32x8;
typedef __attribute__((ext_vector_type(4)))  int int32x4;

constexpr int kB = 32;      // batch
constexpr int kT = 4096;    // time steps
constexpr int kI = 512;     // input dim
constexpr int kS = 512;     // state dim
constexpr int kO = 512;     // output dim
constexpr int kM = kB * kT; // rows of the bx GEMM

// LDS layout for the scan kernel (dynamic LDS, base offset 0):
constexpr unsigned kStateBytes = kB * kS * sizeof(__bf16);  // 32768
constexpr unsigned kLdsState0  = 0;
constexpr unsigned kLdsState1  = kStateBytes;
constexpr unsigned kLdsBx0     = 2 * kStateBytes;
constexpr unsigned kLdsBx1     = 3 * kStateBytes;
constexpr unsigned kLdsTotal   = 4 * kStateBytes;           // 131072 (< 320KB/WGP)

#if __has_builtin(__builtin_amdgcn_tensor_load_to_lds) && \
    __has_builtin(__builtin_amdgcn_s_wait_tensorcnt)
#define HAVE_TDM 1
#endif

// --- WMMA operand loaders (CDNA5 16-bit layouts, see CDNA5 ISA §7.12.2) ---

// A-matrix 16x32 bf16 tile. lane<16: M=lane, K = {k0..k0+7, k0+16..k0+23};
// lane>=16: M=lane-16, K = {k0+8..k0+15, k0+24..k0+31}.
__device__ __forceinline__ v16bf load_a_bf16(const __bf16* p, int ld,
                                             int row0, int k0, int lane) {
  const int r  = row0 + (lane & 15);
  const int kb = k0 + ((lane >> 4) << 3);
  const __bf16* base = p + (size_t)r * ld;
  bf16x8 lo = *reinterpret_cast<const bf16x8*>(base + kb);
  bf16x8 hi = *reinterpret_cast<const bf16x8*>(base + kb + 16);
  v16bf v;
#pragma unroll
  for (int i = 0; i < 8; ++i) { v[i] = lo[i]; v[i + 8] = hi[i]; }
  return v;
}

// Same A layout but source is fp32 (converted to bf16 in-register).
__device__ __forceinline__ v16bf load_a_f32(const float* p, int ld,
                                            int row0, int k0, int lane) {
  const int r  = row0 + (lane & 15);
  const int kb = k0 + ((lane >> 4) << 3);
  const float* base = p + (size_t)r * ld;
  float4 a0 = *reinterpret_cast<const float4*>(base + kb);
  float4 a1 = *reinterpret_cast<const float4*>(base + kb + 4);
  float4 b0 = *reinterpret_cast<const float4*>(base + kb + 16);
  float4 b1 = *reinterpret_cast<const float4*>(base + kb + 20);
  v16bf v;
  v[0]  = (__bf16)a0.x; v[1]  = (__bf16)a0.y; v[2]  = (__bf16)a0.z; v[3]  = (__bf16)a0.w;
  v[4]  = (__bf16)a1.x; v[5]  = (__bf16)a1.y; v[6]  = (__bf16)a1.z; v[7]  = (__bf16)a1.w;
  v[8]  = (__bf16)b0.x; v[9]  = (__bf16)b0.y; v[10] = (__bf16)b0.z; v[11] = (__bf16)b0.w;
  v[12] = (__bf16)b1.x; v[13] = (__bf16)b1.y; v[14] = (__bf16)b1.z; v[15] = (__bf16)b1.w;
  return v;
}

// B-matrix 32x16 bf16 tile, sourced from a row-major weight W[n,k]
// (B[k][n] = W[n][k]; column n of B is row n of W, contiguous in k).
// lane<16: N=lane, K = {k0..k0+7, k0+16..k0+23}; lane>=16: N=lane-16, K = {k0+8..k0+15, k0+24..k0+31}.
__device__ __forceinline__ v16bf load_b_bf16(const __bf16* p, int ld,
                                             int n0, int k0, int lane) {
  const int n  = n0 + (lane & 15);
  const int kb = k0 + ((lane >> 4) << 3);
  const __bf16* base = p + (size_t)n * ld + kb;
  bf16x8 lo = *reinterpret_cast<const bf16x8*>(base);
  bf16x8 hi = *reinterpret_cast<const bf16x8*>(base + 16);
  v16bf v;
#pragma unroll
  for (int i = 0; i < 8; ++i) { v[i] = lo[i]; v[i + 8] = hi[i]; }
  return v;
}

__device__ __forceinline__ v8f wmma_bf16(v16bf a, v16bf b, v8f c) {
  return __builtin_amdgcn_wmma_f32_16x16x32_bf16(
      /*neg_a=*/false, a, /*neg_b=*/false, b,
      /*c_mod=*/(short)0, c, /*reuse_a=*/false, /*reuse_b=*/false);
}

#if defined(HAVE_TDM)
// Issue a TDM 2D tile load: bx slice for time step t (32 rows x 512 bf16,
// row stride kT*kS elements) into LDS at byte offset lds_off.
// D# bitfields per CDNA5 ISA.
__device__ __forceinline__ void tdm_load_bx_slice(const __bf16* bx, int t,
                                                  unsigned lds_off) {
  const uint64_t ga =
      (uint64_t)(uintptr_t)bx + (uint64_t)t * (kS * sizeof(__bf16));
  uint32x4 g0;
  g0[0] = 1u;                                     // count=1, user mode
  g0[1] = lds_off;                                // lds_addr (bytes)
  g0[2] = (unsigned)(ga & 0xffffffffu);           // global_addr[31:0]
  g0[3] = (unsigned)((ga >> 32) & 0x01ffffffu)    // global_addr[56:32]
          | (2u << 30);                           // type = 2 ("image")
  int32x8 g1;
  g1[0] = (1 << 16);                 // data_size = 1 -> 2 bytes/element
  g1[1] = (int)(512u << 16);         // tensor_dim0 = 512
  g1[2] = 0;                         // tensor_dim0 hi, tensor_dim1 lo16 (131072 -> 0)
  g1[3] = (int)(2u | (512u << 16));  // tensor_dim1 = 131072, tile_dim0 = 512
  g1[4] = 32;                        // tile_dim1 = 32 rows, tile_dim2 = 0
  g1[5] = 0x200000;                  // tensor_dim0_stride = kT*kS = 2097152
  g1[6] = 0;
  g1[7] = 0;
  int32x4 gz; gz[0] = 0; gz[1] = 0; gz[2] = 0; gz[3] = 0;
#if defined(__clang_major__) && __clang_major__ >= 23
  int32x8 gz8;
#pragma unroll
  for (int i = 0; i < 8; ++i) gz8[i] = 0;
  __builtin_amdgcn_tensor_load_to_lds(g0, g1, gz, gz, gz8, 0);
#else
  __builtin_amdgcn_tensor_load_to_lds(g0, g1, gz, gz, 0);
#endif
}
#endif  // HAVE_TDM

// --- Kernel 0: convert the three 512x512 fp32 weights to bf16 ----------------
__global__ __launch_bounds__(256) void cvt_kernel(
    const float* __restrict__ Aw, const float* __restrict__ Bw,
    const float* __restrict__ Cw, __bf16* __restrict__ wA,
    __bf16* __restrict__ wB, __bf16* __restrict__ wC) {
  constexpr int N = kS * kS;
  int i = blockIdx.x * 256 + threadIdx.x;
  if (i < N)            wA[i]         = (__bf16)Aw[i];
  else if (i < 2 * N)   wB[i - N]     = (__bf16)Bw[i - N];
  else if (i < 3 * N)   wC[i - 2 * N] = (__bf16)Cw[i - 2 * N];
}

// --- Kernel 1: bx[m,n] = sum_k x[m,k]*B_w[n,k] + B_b[n], stored bf16 ---------
// Per k-step, all 8 B-tiles go to distinct registers so the compiler can
// clause the 16 b128 loads and overlap their waits with the WMMAs.
__global__ __launch_bounds__(256) void bx_gemm_kernel(
    const float* __restrict__ x, const __bf16* __restrict__ wB,
    const float* __restrict__ Bb, __bf16* __restrict__ bx) {
  const int lane = threadIdx.x & 31;
  const int wave = threadIdx.x >> 5;
  const int m0 = blockIdx.x * 128 + wave * 16;
  const int n0 = blockIdx.y * 128;

  v8f acc[8];
#pragma unroll
  for (int jt = 0; jt < 8; ++jt)
#pragma unroll
    for (int r = 0; r < 8; ++r) acc[jt][r] = 0.0f;

  for (int kb = 0; kb < kI; kb += 32) {
    v16bf a = load_a_f32(x, kI, m0, kb, lane);
    v16bf bt[8];
#pragma unroll
    for (int jt = 0; jt < 8; ++jt)
      bt[jt] = load_b_bf16(wB, kI, n0 + jt * 16, kb, lane);
#pragma unroll
    for (int jt = 0; jt < 8; ++jt)
      acc[jt] = wmma_bf16(a, bt[jt], acc[jt]);
  }

#pragma unroll
  for (int jt = 0; jt < 8; ++jt) {
    const int n = n0 + jt * 16 + (lane & 15);
    const float bias = Bb[n];
#pragma unroll
    for (int r = 0; r < 8; ++r) {
      const int m = m0 + r + ((lane >> 4) << 3);
      bx[(size_t)m * kS + n] = (__bf16)(acc[jt][r] + bias);
    }
  }
}

// --- Kernel 2: sequential scan, single workgroup (16 waves on one WGP) -------
// Each wave owns a unique 32-column slice of the new state (2 N-tiles x both
// M-tiles). Its A_w B-operands (2 x 16 v16bf = 256 VGPRs) are loaded ONCE and
// stay register-resident for all 4096 steps, so the per-step critical path is
// only LDS A-loads + WMMA issue. bx_t is TDM double-buffered in LDS.
__global__ __launch_bounds__(512) void scan_kernel(
    const __bf16* __restrict__ wA, const float* __restrict__ Ab,
    const __bf16* __restrict__ bx, float* __restrict__ fs) {
  extern __shared__ __align__(16) char smem[];
  __bf16* st0 = reinterpret_cast<__bf16*>(smem + kLdsState0);
  __bf16* st1 = reinterpret_cast<__bf16*>(smem + kLdsState1);
#if defined(HAVE_TDM)
  __bf16* bxl0 = reinterpret_cast<__bf16*>(smem + kLdsBx0);
  __bf16* bxl1 = reinterpret_cast<__bf16*>(smem + kLdsBx1);
#endif

  const int lane  = threadIdx.x & 31;
  const int wave  = threadIdx.x >> 5;
  const int nbase = wave * 32;          // unique 32 columns per wave

  // t-invariant: weight tiles into registers, biases into registers.
  v16bf bw[2][16];
#pragma unroll
  for (int jn = 0; jn < 2; ++jn)
#pragma unroll
    for (int kb = 0; kb < 16; ++kb)
      bw[jn][kb] = load_b_bf16(wA, kS, nbase + jn * 16, kb * 32, lane);
  const float bias0 = Ab[nbase + (lane & 15)];
  const float bias1 = Ab[nbase + 16 + (lane & 15)];

#if defined(HAVE_TDM)
  if (wave == 0) tdm_load_bx_slice(bx, 0, kLdsBx0);  // preload bx_0
#endif
  for (int i = threadIdx.x; i < kB * kS; i += blockDim.x) st0[i] = (__bf16)0.0f;
#if defined(HAVE_TDM)
  __builtin_amdgcn_s_wait_tensorcnt(0);
#endif
  __syncthreads();

  for (int t = 0; t < kT; ++t) {
    const __bf16* cur = (t & 1) ? st1 : st0;
    __bf16*       nxt = (t & 1) ? st0 : st1;
#if defined(HAVE_TDM)
    const __bf16* bxt = (t & 1) ? bxl1 : bxl0;
    // Kick off the TDM fetch of bx_{t+1}; it runs asynchronously under
    // TENSORcnt while this step's WMMAs execute.
    if (wave == 0 && t + 1 < kT)
      tdm_load_bx_slice(bx, t + 1, ((t + 1) & 1) ? kLdsBx1 : kLdsBx0);
#endif

    v8f a00, a01, a10, a11;
#pragma unroll
    for (int r = 0; r < 8; ++r) { a00[r] = 0.f; a01[r] = 0.f; a10[r] = 0.f; a11[r] = 0.f; }

#pragma unroll
    for (int kb = 0; kb < 16; ++kb) {
      v16bf s0 = load_a_bf16(cur, kS, 0,  kb * 32, lane);
      v16bf s1 = load_a_bf16(cur, kS, 16, kb * 32, lane);
      a00 = wmma_bf16(s0, bw[0][kb], a00);
      a01 = wmma_bf16(s0, bw[1][kb], a01);
      a10 = wmma_bf16(s1, bw[0][kb], a10);
      a11 = wmma_bf16(s1, bw[1][kb], a11);
    }

#pragma unroll
    for (int jm = 0; jm < 2; ++jm)
#pragma unroll
      for (int jn = 0; jn < 2; ++jn) {
        const v8f& acc = jm == 0 ? (jn == 0 ? a00 : a01) : (jn == 0 ? a10 : a11);
        const int n = nbase + jn * 16 + (lane & 15);
        const float bias = jn == 0 ? bias0 : bias1;
#pragma unroll
        for (int r = 0; r < 8; ++r) {
          const int m = jm * 16 + r + ((lane >> 4) << 3);
#if defined(HAVE_TDM)
          float v = acc[r] + bias + (float)bxt[m * kS + n];
#else
          float v = acc[r] + bias + (float)bx[((size_t)m * kT + t) * kS + n];
#endif
          v = fmaxf(v, 0.0f);
          nxt[m * kS + n] = (__bf16)v;
          if (t == kT - 1) fs[m * kS + n] = v;  // fp32 final state for C GEMM
        }
      }
#if defined(HAVE_TDM)
    // Drain the issuing wave's TDM op; the barrier then publishes the new
    // state and the prefetched bx_{t+1} buffer to all waves.
    __builtin_amdgcn_s_wait_tensorcnt(0);
#endif
    __syncthreads();
  }
}

// --- Kernel 3: out = fs @ C_w^T + C_b (fp32 out) -----------------------------
__global__ __launch_bounds__(512) void out_gemm_kernel(
    const float* __restrict__ fs, const __bf16* __restrict__ wC,
    const float* __restrict__ Cb, float* __restrict__ out) {
  const int lane  = threadIdx.x & 31;
  const int wave  = threadIdx.x >> 5;
  const int m0    = (wave >> 3) * 16;
  const int nbase = (wave & 7) * 64;

  v8f acc[4];
#pragma unroll
  for (int jt = 0; jt < 4; ++jt)
#pragma unroll
    for (int r = 0; r < 8; ++r) acc[jt][r] = 0.0f;

  for (int kb = 0; kb < kS; kb += 32) {
    v16bf a = load_a_f32(fs, kS, m0, kb, lane);
#pragma unroll
    for (int jt = 0; jt < 4; ++jt) {
      v16bf b = load_b_bf16(wC, kS, nbase + jt * 16, kb, lane);
      acc[jt] = wmma_bf16(a, b, acc[jt]);
    }
  }

#pragma unroll
  for (int jt = 0; jt < 4; ++jt) {
    const int n = nbase + jt * 16 + (lane & 15);
    const float bias = Cb[n];
#pragma unroll
    for (int r = 0; r < 8; ++r) {
      const int m = m0 + r + ((lane >> 4) << 3);
      out[(size_t)m * kO + n] = acc[jt][r] + bias;
    }
  }
}

// ---------------------------------------------------------------------------
extern "C" void kernel_launch(void* const* d_in, const int* in_sizes, int n_in,
                              void* d_out, int out_size, void* d_ws,
                              size_t ws_size, hipStream_t stream) {
  const float* x  = (const float*)d_in[0];
  const float* Aw = (const float*)d_in[1];
  const float* Ab = (const float*)d_in[2];
  const float* Bw = (const float*)d_in[3];
  const float* Bb = (const float*)d_in[4];
  const float* Cw = (const float*)d_in[5];
  const float* Cb = (const float*)d_in[6];
  float* out = (float*)d_out;

  // Workspace layout: bx(bf16, 134MB) | A_w bf16 | B_w bf16 | C_w bf16 | fs f32
  char* ws = (char*)d_ws;
  size_t off = 0;
  __bf16* bx = (__bf16*)(ws + off); off += (size_t)kM * kS * sizeof(__bf16);
  __bf16* wA = (__bf16*)(ws + off); off += (size_t)kS * kS * sizeof(__bf16);
  __bf16* wB = (__bf16*)(ws + off); off += (size_t)kS * kS * sizeof(__bf16);
  __bf16* wC = (__bf16*)(ws + off); off += (size_t)kS * kS * sizeof(__bf16);
  float*  fs = (float*)(ws + off);  off += (size_t)kB * kS * sizeof(float);
  (void)ws_size; (void)in_sizes; (void)n_in; (void)out_size;

  // Allow 128KB dynamic LDS for the scan kernel (default cap is 64KB).
  (void)hipFuncSetAttribute(reinterpret_cast<const void*>(scan_kernel),
                            hipFuncAttributeMaxDynamicSharedMemorySize,
                            (int)kLdsTotal);

  cvt_kernel<<<(3 * kS * kS) / 256, 256, 0, stream>>>(Aw, Bw, Cw, wA, wB, wC);
  bx_gemm_kernel<<<dim3(kM / 128, kS / 128), 256, 0, stream>>>(x, wB, Bb, bx);
  scan_kernel<<<1, 512, kLdsTotal, stream>>>(wA, Ab, bx, fs);
  out_gemm_kernel<<<1, 512, 0, stream>>>(fs, wC, Cb, out);
}
